// MultiheadAttLayer_42210938585262
// MI455X (gfx1250) — matrix operand order repaired and checked by reading, not compiled, not run on hardware
//
#include <hip/hip_runtime.h>
#include <hip/hip_bf16.h>

typedef __bf16 bf16;
typedef __attribute__((ext_vector_type(16))) __bf16 v16bf;
typedef __attribute__((ext_vector_type(8)))  float  v8f;

namespace {
constexpr int kB  = 4;
constexpr int kL  = 2048;
constexpr int kD  = 1024;
constexpr int kH  = 16;
constexpr int kDK = 64;     // == DV
constexpr int kBL = kB * kL;          // 8192
constexpr int kBH = kB * kH;          // 64
} // namespace

// ---------------------------------------------------------------------------
// Fragment plumbing (wave32, v_wmma_f32_16x16x32_bf16)
//   A: 16x32 bf16, row-major in LDS (pitch multiple of 8) -> 2x ds_load_b128.
//   B: 32x16 bf16, pre-swizzled fragment blocks in LDS    -> 2x ds_load_b128.
//   C/D: 8xf32. M = vgpr + (lane>>4)*8, N = lane&15.
// ---------------------------------------------------------------------------
union Frag {
  v16bf v;
  uint4 q[2];
};

__device__ inline v16bf load_a_frag(const bf16* a, int lda) {
  int lane = threadIdx.x & 31;
  int hf = lane >> 4, row = lane & 15;
  const bf16* p = a + row * lda + hf * 16;
  Frag f;
  f.q[0] = *(const uint4*)(p);
  f.q[1] = *(const uint4*)(p + 8);
  return f.v;
}

__device__ inline v16bf load_b_frag(const bf16* blockBase) {
  int lane = threadIdx.x & 31;
  const bf16* p = blockBase + lane * 16;
  Frag f;
  f.q[0] = *(const uint4*)(p);
  f.q[1] = *(const uint4*)(p + 8);
  return f.v;
}

__device__ inline int b_sw_off(int k, int n) {
  int block = n >> 4;
  int lane  = ((k >> 4) << 4) | (n & 15);
  return (block * 32 + lane) * 16 + (k & 15);
}

__device__ inline v8f wmma_bf16(v16bf a, v16bf b, v8f c) {
  return __builtin_amdgcn_wmma_f32_16x16x32_bf16(
      false, a, false, b, (short)0, c, false, false);
}

__device__ inline unsigned pack_bf16x2(float lo, float hi) {
  union { bf16 b[2]; unsigned u; } pk;
  pk.b[0] = (bf16)lo;
  pk.b[1] = (bf16)hi;
  return pk.u;
}

// ---------------------------------------------------------------------------
// Kernel 1: QKV projection GEMM, double-buffered.
// C[8192,1024] = X(f32->bf16) @ W(f32->bf16) + bias. Block tile 128x64.
// transK==0 -> [B,H,L,64]; transK==1 -> [B,H,64,L].
// ---------------------------------------------------------------------------
__global__ __launch_bounds__(256) void proj_gemm_kernel(
    const float* __restrict__ X, const float* __restrict__ W,
    const float* __restrict__ bias, bf16* __restrict__ out,
    float scale, int transK) {
  constexpr int LDA = 40;
  __shared__ alignas(16) bf16 sA[2][128 * LDA];
  __shared__ alignas(16) bf16 sB[2][4 * 512];

  const int tid = threadIdx.x;
  const int w = tid >> 5, lane = tid & 31;
  const int hf = lane >> 4, col = lane & 15;
  const int mBlock = blockIdx.y * 128;
  const int nBlock = blockIdx.x * 64;

  // Per-thread staging coordinates (8 A float2's, 8 B floats per k-step).
  int aR[8], aC[8], bR[8], bC[8];
#pragma unroll
  for (int j = 0; j < 8; ++j) {
    int ia = (tid + j * 256) * 2;       // over 128x32 A tile
    aR[j] = ia >> 5; aC[j] = ia & 31;
    int ib = tid + j * 256;             // over 32x64 B tile
    bR[j] = ib >> 6; bC[j] = ib & 63;
  }

  float2 pa[8];
  float  pb[8];
  // ---- stage k0 = 0
#pragma unroll
  for (int j = 0; j < 8; ++j) {
    pa[j] = *(const float2*)&X[(size_t)(mBlock + aR[j]) * kD + aC[j]];
    pb[j] = W[(size_t)bR[j] * kD + nBlock + bC[j]];
  }
#pragma unroll
  for (int j = 0; j < 8; ++j) {
    *(unsigned*)&sA[0][aR[j] * LDA + aC[j]] = pack_bf16x2(pa[j].x, pa[j].y);
    sB[0][b_sw_off(bR[j], bC[j])] = (bf16)pb[j];
  }
  __syncthreads();

  v8f acc[4] = {};
  int cur = 0;
  for (int k0 = 0; k0 < kD; k0 += 32) {
    const int kn = k0 + 32;
    const bool more = kn < kD;
    if (more) {
#pragma unroll
      for (int j = 0; j < 8; ++j) {
        pa[j] = *(const float2*)&X[(size_t)(mBlock + aR[j]) * kD + kn + aC[j]];
        pb[j] = W[(size_t)(kn + bR[j]) * kD + nBlock + bC[j]];
      }
    }
    // compute current buffer (batch all fragment loads before the WMMA chain)
    v16bf af = load_a_frag(&sA[cur][(w * 16) * LDA], LDA);
    v16bf b0 = load_b_frag(&sB[cur][0 * 512]);
    v16bf b1 = load_b_frag(&sB[cur][1 * 512]);
    v16bf b2 = load_b_frag(&sB[cur][2 * 512]);
    v16bf b3 = load_b_frag(&sB[cur][3 * 512]);
    acc[0] = wmma_bf16(af, b0, acc[0]);
    acc[1] = wmma_bf16(af, b1, acc[1]);
    acc[2] = wmma_bf16(af, b2, acc[2]);
    acc[3] = wmma_bf16(af, b3, acc[3]);
    if (more) {
#pragma unroll
      for (int j = 0; j < 8; ++j) {
        *(unsigned*)&sA[cur ^ 1][aR[j] * LDA + aC[j]] =
            pack_bf16x2(pa[j].x, pa[j].y);
        sB[cur ^ 1][b_sw_off(bR[j], bC[j])] = (bf16)pb[j];
      }
    }
    __syncthreads();
    cur ^= 1;
  }

#pragma unroll
  for (int t = 0; t < 4; ++t) {
#pragma unroll
    for (int r = 0; r < 8; ++r) {
      int m = mBlock + w * 16 + r + hf * 8;
      int n = nBlock + t * 16 + col;
      float vv = (acc[t][r] + bias[n]) * scale;
      int b = m >> 11, l = m & (kL - 1);
      int h = n >> 6, dk = n & 63;
      size_t idx = transK
          ? (((size_t)(b * kH + h) * kDK + dk) * kL + l)
          : (((size_t)(b * kH + h) * kL + l) * kDK + dk);
      out[idx] = (bf16)vv;
    }
  }
}

// ---------------------------------------------------------------------------
// Kernel 2: scores. Per (b,h): logits[2048,2048] = qh_s @ khT, masked.
// Block tile 128x128, K=64 (two WMMA k-steps, tiles loaded once).
// ---------------------------------------------------------------------------
__global__ __launch_bounds__(256) void scores_kernel(
    const bf16* __restrict__ qh, const bf16* __restrict__ khT,
    const int* __restrict__ mask, float* __restrict__ att) {
  constexpr int LDQ = 72;
  __shared__ alignas(16) bf16 sQ[128 * LDQ];
  __shared__ alignas(16) bf16 sK[16 * 512];

  const int tid = threadIdx.x;
  const int w = tid >> 5, lane = tid & 31;
  const int hf = lane >> 4, col = lane & 15;
  const int bh = blockIdx.z;
  const int b = bh / kH;
  const int mBlock = blockIdx.y * 128;
  const int nBlock = blockIdx.x * 128;

  const bf16* qb = qh  + (size_t)bh * kL * kDK;
  const bf16* kb = khT + (size_t)bh * kDK * kL;

  for (int j = tid; j < 2048; j += 256) {
    int idx = j * 4;
    int r = idx >> 6, c = idx & 63;
    *(uint2*)&sQ[r * LDQ + c] =
        *(const uint2*)&qb[(size_t)(mBlock + r) * kDK + c];
  }
  for (int i = tid; i < 8192; i += 256) {
    int r = i >> 7, c = i & 127;                 // r = dk, c = key
    int blockIx = (r >> 5) * 8 + (c >> 4);
    int lfrag = (((r >> 4) & 1) << 4) | (c & 15);
    sK[(blockIx * 32 + lfrag) * 16 + (r & 15)] = kb[(size_t)r * kL + nBlock + c];
  }
  __syncthreads();

  v8f acc[8] = {};
#pragma unroll
  for (int ks = 0; ks < 2; ++ks) {
    v16bf af = load_a_frag(&sQ[(w * 16) * LDQ + ks * 32], LDQ);
#pragma unroll
    for (int g = 0; g < 2; ++g) {     // two groups of 4 tiles
      v16bf b0 = load_b_frag(&sK[(ks * 8 + g * 4 + 0) * 512]);
      v16bf b1 = load_b_frag(&sK[(ks * 8 + g * 4 + 1) * 512]);
      v16bf b2 = load_b_frag(&sK[(ks * 8 + g * 4 + 2) * 512]);
      v16bf b3 = load_b_frag(&sK[(ks * 8 + g * 4 + 3) * 512]);
      acc[g * 4 + 0] = wmma_bf16(af, b0, acc[g * 4 + 0]);
      acc[g * 4 + 1] = wmma_bf16(af, b1, acc[g * 4 + 1]);
      acc[g * 4 + 2] = wmma_bf16(af, b2, acc[g * 4 + 2]);
      acc[g * 4 + 3] = wmma_bf16(af, b3, acc[g * 4 + 3]);
    }
  }

#pragma unroll
  for (int t = 0; t < 8; ++t) {
#pragma unroll
    for (int r = 0; r < 8; ++r) {
      int m = mBlock + w * 16 + r + hf * 8;
      int n = nBlock + t * 16 + col;
      int mk = mask[((size_t)b * kL + m) * kL + n];
      att[((size_t)bh * kL + m) * kL + n] = mk ? acc[t][r] : -1e9f;
    }
  }
}

// ---------------------------------------------------------------------------
// Kernel 3: per-row softmax stats. One wave per 2048-wide row.
// ---------------------------------------------------------------------------
__global__ __launch_bounds__(256) void softmax_stats_kernel(
    const float* __restrict__ att, float* __restrict__ stats) {
  int w = threadIdx.x >> 5, lane = threadIdx.x & 31;
  size_t row = (size_t)blockIdx.x * 8 + w;
  const float* p = att + row * kL;

  float vals[64];
  float mx = -3.0e38f;
#pragma unroll
  for (int i = 0; i < 64; ++i) {
    vals[i] = p[lane + i * 32];
    mx = fmaxf(mx, vals[i]);
  }
#pragma unroll
  for (int off = 16; off > 0; off >>= 1)
    mx = fmaxf(mx, __shfl_xor(mx, off, 32));

  float s = 0.f;
#pragma unroll
  for (int i = 0; i < 64; ++i) s += __expf(vals[i] - mx);
#pragma unroll
  for (int off = 16; off > 0; off >>= 1)
    s += __shfl_xor(s, off, 32);

  if (lane == 0) {
    stats[row * 2]     = mx;
    stats[row * 2 + 1] = 1.0f / s;
  }
}

// ---------------------------------------------------------------------------
// Kernel 4: attend, double-buffered. Normalizes logits -> probs (written back
// to d_out att in-place as each tile is staged) and WMMAs probs against vh.
// Block tile 128 queries x 64 dv; softmax stats hoisted out of the k-loop.
// ---------------------------------------------------------------------------
__global__ __launch_bounds__(256) void attend_kernel(
    float* __restrict__ att, const float* __restrict__ stats,
    const bf16* __restrict__ vh, bf16* __restrict__ attout) {
  constexpr int LDP = 40;
  __shared__ alignas(16) bf16 sP[2][128 * LDP];
  __shared__ alignas(16) bf16 sV[2][4 * 512];

  const int tid = threadIdx.x;
  const int w = tid >> 5, lane = tid & 31;
  const int hf = lane >> 4, col = lane & 15;
  const int bh = blockIdx.y;
  const int b = bh / kH, h = bh % kH;
  const int mBlock = blockIdx.x * 128;

  const bf16* vb = vh + (size_t)bh * kL * kDK;

  // Staging coordinates: 8 float2's over 128x32 probs tile, 8 bf16 over 32x64 V.
  int pR[8], pC[8], vR[8], vC[8];
  float mxv[8], inv[8];
  size_t prow[8];
#pragma unroll
  for (int j = 0; j < 8; ++j) {
    int ip = (tid + j * 256) * 2;
    pR[j] = ip >> 5; pC[j] = ip & 31;
    int iv = tid + j * 256;
    vR[j] = iv >> 6; vC[j] = iv & 63;
    prow[j] = (size_t)bh * kL + mBlock + pR[j];
    mxv[j] = stats[prow[j] * 2];        // loop-invariant softmax stats
    inv[j] = stats[prow[j] * 2 + 1];
  }

  float2 px[8];
  bf16   pv[8];
  // ---- stage k0 = 0
#pragma unroll
  for (int j = 0; j < 8; ++j) {
    px[j] = *(const float2*)&att[prow[j] * kL + pC[j]];
    pv[j] = vb[(size_t)vR[j] * kDK + vC[j]];
  }
#pragma unroll
  for (int j = 0; j < 8; ++j) {
    float p0 = __expf(px[j].x - mxv[j]) * inv[j];
    float p1 = __expf(px[j].y - mxv[j]) * inv[j];
    *(float2*)&att[prow[j] * kL + pC[j]] = make_float2(p0, p1);
    *(unsigned*)&sP[0][pR[j] * LDP + pC[j]] = pack_bf16x2(p0, p1);
    sV[0][b_sw_off(vR[j], vC[j])] = pv[j];
  }
  __syncthreads();

  v8f acc[4] = {};
  int cur = 0;
  for (int k0 = 0; k0 < kL; k0 += 32) {
    const int kn = k0 + 32;
    const bool more = kn < kL;
    if (more) {
#pragma unroll
      for (int j = 0; j < 8; ++j) {
        px[j] = *(const float2*)&att[prow[j] * kL + kn + pC[j]];
        pv[j] = vb[(size_t)(kn + vR[j]) * kDK + vC[j]];
      }
    }
    v16bf af = load_a_frag(&sP[cur][(w * 16) * LDP], LDP);
    v16bf b0 = load_b_frag(&sV[cur][0 * 512]);
    v16bf b1 = load_b_frag(&sV[cur][1 * 512]);
    v16bf b2 = load_b_frag(&sV[cur][2 * 512]);
    v16bf b3 = load_b_frag(&sV[cur][3 * 512]);
    acc[0] = wmma_bf16(af, b0, acc[0]);
    acc[1] = wmma_bf16(af, b1, acc[1]);
    acc[2] = wmma_bf16(af, b2, acc[2]);
    acc[3] = wmma_bf16(af, b3, acc[3]);
    if (more) {
#pragma unroll
      for (int j = 0; j < 8; ++j) {
        float p0 = __expf(px[j].x - mxv[j]) * inv[j];
        float p1 = __expf(px[j].y - mxv[j]) * inv[j];
        *(float2*)&att[prow[j] * kL + kn + pC[j]] = make_float2(p0, p1);
        *(unsigned*)&sP[cur ^ 1][pR[j] * LDP + pC[j]] = pack_bf16x2(p0, p1);
        sV[cur ^ 1][b_sw_off(vR[j], vC[j])] = pv[j];
      }
    }
    __syncthreads();
    cur ^= 1;
  }

#pragma unroll
  for (int t = 0; t < 4; ++t) {
#pragma unroll
    for (int r = 0; r < 8; ++r) {
      int m = mBlock + w * 16 + r + hf * 8;
      int n = t * 16 + col;
      attout[((size_t)(b * kL + m)) * kD + h * kDK + n] = (bf16)acc[t][r];
    }
  }
}

// ---------------------------------------------------------------------------
// Kernel 5: output FC + residual, double-buffered.
// out[8192,1024] = attout(bf16) @ Wfc(f32->bf16) + bfc + q, f32.
// ---------------------------------------------------------------------------
__global__ __launch_bounds__(256) void fc_gemm_kernel(
    const bf16* __restrict__ A, const float* __restrict__ W,
    const float* __restrict__ bias, const float* __restrict__ res,
    float* __restrict__ out) {
  constexpr int LDA = 40;
  __shared__ alignas(16) bf16 sA[2][128 * LDA];
  __shared__ alignas(16) bf16 sB[2][4 * 512];

  const int tid = threadIdx.x;
  const int w = tid >> 5, lane = tid & 31;
  const int hf = lane >> 4, col = lane & 15;
  const int mBlock = blockIdx.y * 128;
  const int nBlock = blockIdx.x * 64;

  int aR[4], aC[4], bR[8], bC[8];
#pragma unroll
  for (int j = 0; j < 4; ++j) {
    int ia = (tid + j * 256) * 4;       // 4 bf16 per copy, 128x32 tile
    aR[j] = ia >> 5; aC[j] = ia & 31;
  }
#pragma unroll
  for (int j = 0; j < 8; ++j) {
    int ib = tid + j * 256;             // 32x64 B tile
    bR[j] = ib >> 6; bC[j] = ib & 63;
  }

  uint2 pa[4];
  float pb[8];
#pragma unroll
  for (int j = 0; j < 4; ++j)
    pa[j] = *(const uint2*)&A[(size_t)(mBlock + aR[j]) * kD + aC[j]];
#pragma unroll
  for (int j = 0; j < 8; ++j)
    pb[j] = W[(size_t)bR[j] * kD + nBlock + bC[j]];
#pragma unroll
  for (int j = 0; j < 4; ++j)
    *(uint2*)&sA[0][aR[j] * LDA + aC[j]] = pa[j];
#pragma unroll
  for (int j = 0; j < 8; ++j)
    sB[0][b_sw_off(bR[j], bC[j])] = (bf16)pb[j];
  __syncthreads();

  v8f acc[4] = {};
  int cur = 0;
  for (int k0 = 0; k0 < kD; k0 += 32) {
    const int kn = k0 + 32;
    const bool more = kn < kD;
    if (more) {
#pragma unroll
      for (int j = 0; j < 4; ++j)
        pa[j] = *(const uint2*)&A[(size_t)(mBlock + aR[j]) * kD + kn + aC[j]];
#pragma unroll
      for (int j = 0; j < 8; ++j)
        pb[j] = W[(size_t)(kn + bR[j]) * kD + nBlock + bC[j]];
    }
    v16bf af = load_a_frag(&sA[cur][(w * 16) * LDA], LDA);
    v16bf b0 = load_b_frag(&sB[cur][0 * 512]);
    v16bf b1 = load_b_frag(&sB[cur][1 * 512]);
    v16bf b2 = load_b_frag(&sB[cur][2 * 512]);
    v16bf b3 = load_b_frag(&sB[cur][3 * 512]);
    acc[0] = wmma_bf16(af, b0, acc[0]);
    acc[1] = wmma_bf16(af, b1, acc[1]);
    acc[2] = wmma_bf16(af, b2, acc[2]);
    acc[3] = wmma_bf16(af, b3, acc[3]);
    if (more) {
#pragma unroll
      for (int j = 0; j < 4; ++j)
        *(uint2*)&sA[cur ^ 1][aR[j] * LDA + aC[j]] = pa[j];
#pragma unroll
      for (int j = 0; j < 8; ++j)
        sB[cur ^ 1][b_sw_off(bR[j], bC[j])] = (bf16)pb[j];
    }
    __syncthreads();
    cur ^= 1;
  }

#pragma unroll
  for (int t = 0; t < 4; ++t) {
#pragma unroll
    for (int r = 0; r < 8; ++r) {
      int m = mBlock + w * 16 + r + hf * 8;
      int n = nBlock + t * 16 + col;
      out[(size_t)m * kD + n] = acc[t][r] + bias[n] + res[(size_t)m * kD + n];
    }
  }
}

// ---------------------------------------------------------------------------
// Host launcher
// ---------------------------------------------------------------------------
extern "C" void kernel_launch(void* const* d_in, const int* in_sizes, int n_in,
                              void* d_out, int out_size, void* d_ws, size_t ws_size,
                              hipStream_t stream) {
  (void)in_sizes; (void)n_in; (void)out_size; (void)ws_size;

  const float* q   = (const float*)d_in[0];
  const float* k   = (const float*)d_in[1];
  const float* v   = (const float*)d_in[2];
  const int*   msk = (const int*)  d_in[3];
  const float* Wq  = (const float*)d_in[4];
  const float* bq  = (const float*)d_in[5];
  const float* Wk  = (const float*)d_in[6];
  const float* bk  = (const float*)d_in[7];
  const float* Wv  = (const float*)d_in[8];
  const float* bv  = (const float*)d_in[9];
  const float* Wfc = (const float*)d_in[10];
  const float* bfc = (const float*)d_in[11];

  float* outp = (float*)d_out;
  float* att  = outp + (size_t)kBL * kD;   // d_out = [out | att]

  const size_t nProj = (size_t)kB * kH * kL * kDK;   // 8,388,608 elements
  bf16* qh     = (bf16*)d_ws;
  bf16* khT    = qh  + nProj;
  bf16* vh     = khT + nProj;
  bf16* attout = vh  + nProj;
  float* stats = (float*)(attout + nProj);           // [B*H*L][2]

  dim3 blk(256);
  proj_gemm_kernel<<<dim3(16, 64), blk, 0, stream>>>(q, Wq, bq, qh, 0.125f, 0);
  proj_gemm_kernel<<<dim3(16, 64), blk, 0, stream>>>(k, Wk, bk, khT, 1.0f, 1);
  proj_gemm_kernel<<<dim3(16, 64), blk, 0, stream>>>(v, Wv, bv, vh, 1.0f, 0);
  scores_kernel<<<dim3(16, 16, kBH), blk, 0, stream>>>(qh, khT, msk, att);
  softmax_stats_kernel<<<dim3((kB * kH * kL) / 8), blk, 0, stream>>>(att, stats);
  attend_kernel<<<dim3(16, kBH), blk, 0, stream>>>(att, stats, vh, attout);
  fc_gemm_kernel<<<dim3(16, 64), blk, 0, stream>>>(attout, Wfc, bfc, q, outp);
}
